// ONE_ATTENTION_11175504904854
// MI455X (gfx1250) — matrix operand order repaired and checked by reading, not compiled, not run on hardware
//
#include <hip/hip_runtime.h>
#include <hip/hip_bf16.h>
#include <math.h>

#define FDIM 768
#define NBATCH 4096
#define NNODE 21
#define NEVID 20

typedef __attribute__((ext_vector_type(4)))  _Float16 h4;
typedef __attribute__((ext_vector_type(8)))  _Float16 h8;
typedef __attribute__((ext_vector_type(16))) _Float16 h16;
typedef __attribute__((ext_vector_type(8)))  float    f8v;

__device__ __forceinline__ float eluf(float x) { return x > 0.f ? x : expm1f(x); }

// ---------------------------------------------------------------------------
// WMMA GEMM: C[M x N] (f16) = A[M x K] (f16, row major) * Bt[N x K]^T (f16)
// optional epilogue: C = elu(C + bias[col]).
// Block: 256 thr = 8 waves. Block tile 128x128, K-stage 64, double-buffered
// LDS fed by global_load_async_to_lds_b128 (ASYNCcnt). Wave tile 32x64.
// Requires M%128==0, N%128==0 (N=128 ok), K%64==0.
// ---------------------------------------------------------------------------
template <bool BIAS_ELU>
__global__ __launch_bounds__(256) void wmma_gemm(
    const _Float16* __restrict__ A, const _Float16* __restrict__ Bt,
    _Float16* __restrict__ C, const float* __restrict__ bias,
    int M, int N, int K)
{
    constexpr int BM = 128, BN = 128, KT = 64;
    __shared__ _Float16 lsA[2][BM * KT];   // 2 x 16 KB
    __shared__ _Float16 lsB[2][BN * KT];   // 2 x 16 KB

    const int tid  = threadIdx.x;
    const int lane = tid & 31;
    const int w    = tid >> 5;
    const int wm   = w >> 1;      // 0..3
    const int wn   = w & 1;       // 0..1
    const int m0   = blockIdx.x * BM;
    const int n0   = blockIdx.y * BN;
    const int lh   = lane & 15;   // M-row / N-col within 16x16 tile
    const int hh   = lane >> 4;   // half-wave select

    // this thread's 4 A-chunks and 4 B-chunks (16B each) per stage
    f8v acc[2][4] = {};

    // issue async global->LDS copies for one K stage into buffer `buf`
    auto issue_stage = [&](int buf, int k0) {
        #pragma unroll
        for (int i = 0; i < 4; i++) {
            int c = tid + 256 * i;
            int r = c >> 3, s = c & 7;
            unsigned la = (unsigned)(uintptr_t)&lsA[buf][r * KT + s * 8];
            unsigned long long ga =
                (unsigned long long)(uintptr_t)&A[(size_t)(m0 + r) * K + k0 + s * 8];
            asm volatile("global_load_async_to_lds_b128 %0, %1, off"
                         :: "v"(la), "v"(ga) : "memory");
        }
        #pragma unroll
        for (int i = 0; i < 4; i++) {
            int c = tid + 256 * i;
            int r = c >> 3, s = c & 7;
            unsigned la = (unsigned)(uintptr_t)&lsB[buf][r * KT + s * 8];
            unsigned long long ga =
                (unsigned long long)(uintptr_t)&Bt[(size_t)(n0 + r) * K + k0 + s * 8];
            asm volatile("global_load_async_to_lds_b128 %0, %1, off"
                         :: "v"(la), "v"(ga) : "memory");
        }
    };

    const int nstage = K / KT;
    issue_stage(0, 0);
    asm volatile("s_wait_asynccnt 0x0" ::: "memory");
    __syncthreads();

    for (int s = 0; s < nstage; s++) {
        const int buf = s & 1;
        if (s + 1 < nstage) issue_stage(buf ^ 1, (s + 1) * KT);  // overlap with compute

        #pragma unroll
        for (int kk = 0; kk < KT; kk += 32) {
            h16 af[2], bf[4];
            // A frag: lane row = lh, 16 contiguous k starting at hh*16
            // (elems k..k+7 then k+8..k+15, same order as B)
            #pragma unroll
            for (int mi = 0; mi < 2; mi++) {
                int r  = wm * 32 + mi * 16 + lh;
                h8 lo = *(const h8*)&lsA[buf][r * KT + kk + hh * 16];
                h8 hi = *(const h8*)&lsA[buf][r * KT + kk + hh * 16 + 8];
                #pragma unroll
                for (int q = 0; q < 8; q++) { af[mi][q] = lo[q]; af[mi][q + 8] = hi[q]; }
            }
            // B frag: lane col = lh, 16 contiguous k starting at hh*16
            #pragma unroll
            for (int ni = 0; ni < 4; ni++) {
                int cc = wn * 64 + ni * 16 + lh;
                bf[ni] = *(const h16*)&lsB[buf][cc * KT + kk + hh * 16];
            }
            #pragma unroll
            for (int mi = 0; mi < 2; mi++)
                #pragma unroll
                for (int ni = 0; ni < 4; ni++)
                    acc[mi][ni] = __builtin_amdgcn_wmma_f32_16x16x32_f16(
                        false, af[mi], false, bf[ni], (short)0, acc[mi][ni], false, false);
        }

        if (s + 1 < nstage) {
            asm volatile("s_wait_asynccnt 0x0" ::: "memory");  // next tile landed
            __syncthreads();                                    // visible to all waves
        }
    }

    // epilogue: D layout -> row = base + hh*8 + q, col = lh
    #pragma unroll
    for (int mi = 0; mi < 2; mi++) {
        int rb = m0 + wm * 32 + mi * 16 + hh * 8;
        #pragma unroll
        for (int ni = 0; ni < 4; ni++) {
            int cc = n0 + wn * 64 + ni * 16 + lh;
            float bv = BIAS_ELU ? bias[cc] : 0.f;
            #pragma unroll
            for (int q = 0; q < 8; q++) {
                float v = acc[mi][ni][q];
                if (BIAS_ELU) v = eluf(v + bv);
                C[(size_t)(rb + q) * N + cc] = (_Float16)v;
            }
        }
    }
}

// ---------------------------------------------------------------------------
// prep kernels
// ---------------------------------------------------------------------------
__global__ __launch_bounds__(256) void kcnt_kernel(const int* __restrict__ lab, int* __restrict__ kc)
{
    int b = blockIdx.x * 256 + threadIdx.x;
    if (b < NBATCH) {
        int s = 0;
        for (int n = 0; n < NNODE; n++) s += lab[b * NNODE + n];
        kc[b] = s;
    }
}

__global__ __launch_bounds__(256) void cvt_f32_h16(const float* __restrict__ in,
                                                   _Float16* __restrict__ out, size_t n4)
{
    size_t i = (size_t)blockIdx.x * 256 + threadIdx.x;
    if (i < n4) {
        float4 v = ((const float4*)in)[i];
        h4 o; o[0] = (_Float16)v.x; o[1] = (_Float16)v.y; o[2] = (_Float16)v.z; o[3] = (_Float16)v.w;
        ((h4*)out)[i] = o;
    }
}

// Wt[n][k] = (f16) W[k][n]   (W is K x N row-major)
__global__ __launch_bounds__(256) void tr_cvt(const float* __restrict__ W,
                                              _Float16* __restrict__ Wt, int K, int N)
{
    int o = blockIdx.x * 256 + threadIdx.x;
    if (o < K * N) {
        int nn = o / K, kk = o % K;
        Wt[o] = (_Float16)W[(size_t)kk * N + nn];
    }
}

// PWt[j][k]: j<64 -> pw1[k][j] (claim half); j>=64 -> pw1[768+k][j-64] (evid half)
__global__ __launch_bounds__(256) void build_pw(const float* __restrict__ pw1,
                                                _Float16* __restrict__ PWt)
{
    int o = blockIdx.x * 256 + threadIdx.x;
    if (o < 128 * FDIM) {
        int j = o / FDIM, k = o % FDIM;
        float v = (j < 64) ? pw1[(size_t)k * 64 + j] : pw1[(size_t)(FDIM + k) * 64 + (j - 64)];
        PWt[o] = (_Float16)v;
    }
}

// ---------------------------------------------------------------------------
// GCN clique aggregation (+bias+elu), one block per batch
// ---------------------------------------------------------------------------
__global__ __launch_bounds__(256) void gcn_agg(const _Float16* __restrict__ xw,
                                               const float* __restrict__ bias,
                                               const int* __restrict__ kc,
                                               _Float16* __restrict__ out)
{
    int b = blockIdx.x;
    const _Float16* xb = xw + (size_t)b * NNODE * FDIM;
    _Float16* ob = out + (size_t)b * NNODE * FDIM;
    int k = kc[b];
    float invdeg = 1.f / (2.f * k + 1.f);
    for (int f = threadIdx.x; f < FDIM; f += 256) {
        float s = 0.f;
        for (int n = 0; n < k; n++) s += (float)xb[n * FDIM + f];
        float bv = bias[f], s2 = 2.f * s;
        for (int n = 0; n < NNODE; n++) {
            float v = (float)xb[n * FDIM + f];
            float a = (n < k) ? (s2 + v) * invdeg : v;
            ob[n * FDIM + f] = (_Float16)eluf(a + bv);
        }
    }
}

// same + row L2-normalize (layer 2)
__global__ __launch_bounds__(256) void gcn_agg_norm(const _Float16* __restrict__ xw,
                                                    const float* __restrict__ bias,
                                                    const int* __restrict__ kc,
                                                    _Float16* __restrict__ out)
{
    __shared__ _Float16 vb[NNODE * FDIM];   // 32.25 KB
    int b = blockIdx.x;
    const _Float16* xb = xw + (size_t)b * NNODE * FDIM;
    int k = kc[b];
    float invdeg = 1.f / (2.f * k + 1.f);
    for (int f = threadIdx.x; f < FDIM; f += 256) {
        float s = 0.f;
        for (int n = 0; n < k; n++) s += (float)xb[n * FDIM + f];
        float bv = bias[f], s2 = 2.f * s;
        for (int n = 0; n < NNODE; n++) {
            float v = (float)xb[n * FDIM + f];
            float a = (n < k) ? (s2 + v) * invdeg : v;
            vb[n * FDIM + f] = (_Float16)eluf(a + bv);
        }
    }
    __syncthreads();
    int w = threadIdx.x >> 5, lane = threadIdx.x & 31;
    _Float16* ob = out + (size_t)b * NNODE * FDIM;
    for (int n = w; n < NNODE; n += 8) {
        float sq = 0.f;
        for (int f = lane; f < FDIM; f += 32) { float v = (float)vb[n * FDIM + f]; sq += v * v; }
        #pragma unroll
        for (int off = 16; off; off >>= 1) sq += __shfl_xor(sq, off, 32);
        float inv = 1.f / fmaxf(sqrtf(sq), 1e-12f);
        for (int f = lane; f < FDIM; f += 32)
            ob[n * FDIM + f] = (_Float16)((float)vb[n * FDIM + f] * inv);
    }
}

// ---------------------------------------------------------------------------
// attention: Z rows = [claim-proj(0:64) | evid-proj(64:128)], per-batch block
// ---------------------------------------------------------------------------
__global__ __launch_bounds__(256) void attn_kernel(const _Float16* __restrict__ Z,
                                                   const _Float16* __restrict__ X2,
                                                   const int* __restrict__ lab,
                                                   const float* __restrict__ pb1,
                                                   const float* __restrict__ pw2,
                                                   const float* __restrict__ pb2,
                                                   _Float16* __restrict__ g)
{
    __shared__ float sl[NEVID];
    __shared__ float wgt[NEVID];
    int b = blockIdx.x;
    const _Float16* zb = Z + (size_t)b * NNODE * 128;
    int w = threadIdx.x >> 5, lane = threadIdx.x & 31;

    for (int e = w; e < NEVID; e += 8) {
        float d = 0.f;
        for (int j = lane; j < 64; j += 32) {
            float u = (float)zb[j] + (float)zb[(1 + e) * 128 + 64 + j] + pb1[j];
            d += eluf(u) * pw2[j];
        }
        #pragma unroll
        for (int off = 16; off; off >>= 1) d += __shfl_xor(d, off, 32);
        if (lane == 0) {
            float s = eluf(d + pb2[0]);
            sl[e] = (lab[b * NNODE + 1 + e] == 0) ? -INFINITY : s;
        }
    }
    __syncthreads();
    if (w == 0) {
        float v = (lane < NEVID) ? sl[lane] : -INFINITY;
        float m = v;
        #pragma unroll
        for (int off = 16; off; off >>= 1) m = fmaxf(m, __shfl_xor(m, off, 32));
        float ex = (lane < NEVID && v > -INFINITY) ? expf(v - m) : 0.f;
        float sum = ex;
        #pragma unroll
        for (int off = 16; off; off >>= 1) sum += __shfl_xor(sum, off, 32);
        if (lane < NEVID) wgt[lane] = ex / sum;
    }
    __syncthreads();
    const _Float16* xb = X2 + (size_t)b * NNODE * FDIM;
    for (int f = threadIdx.x; f < FDIM; f += 256) {
        float a = 0.f;
        for (int e = 0; e < NEVID; e++) a += wgt[e] * (float)xb[(1 + e) * FDIM + f];
        g[(size_t)b * FDIM + f] = (_Float16)a;
    }
}

// ---------------------------------------------------------------------------
// final 768 -> 3 classifier head, one block per batch
// ---------------------------------------------------------------------------
__global__ __launch_bounds__(256) void final_kernel(const _Float16* __restrict__ H,
                                                    const float* __restrict__ cw2,
                                                    const float* __restrict__ cb2,
                                                    float* __restrict__ out)
{
    __shared__ float sw[8][3];
    int b = blockIdx.x;
    float p0 = 0, p1 = 0, p2 = 0;
    for (int f = threadIdx.x; f < FDIM; f += 256) {
        float h = (float)H[(size_t)b * FDIM + f];
        p0 += h * cw2[f * 3 + 0];
        p1 += h * cw2[f * 3 + 1];
        p2 += h * cw2[f * 3 + 2];
    }
    #pragma unroll
    for (int off = 16; off; off >>= 1) {
        p0 += __shfl_xor(p0, off, 32);
        p1 += __shfl_xor(p1, off, 32);
        p2 += __shfl_xor(p2, off, 32);
    }
    int w = threadIdx.x >> 5, lane = threadIdx.x & 31;
    if (lane == 0) { sw[w][0] = p0; sw[w][1] = p1; sw[w][2] = p2; }
    __syncthreads();
    if (threadIdx.x < 3) {
        float s = 0;
        #pragma unroll
        for (int i = 0; i < 8; i++) s += sw[i][threadIdx.x];
        out[(size_t)b * 3 + threadIdx.x] = eluf(s + cb2[threadIdx.x]);
    }
}

// ---------------------------------------------------------------------------
extern "C" void kernel_launch(void* const* d_in, const int* in_sizes, int n_in,
                              void* d_out, int out_size, void* d_ws, size_t ws_size,
                              hipStream_t stream)
{
    (void)in_sizes; (void)n_in; (void)out_size; (void)ws_size;
    const float* pooled = (const float*)d_in[0];
    const float* W1  = (const float*)d_in[1];
    const float* b1  = (const float*)d_in[2];
    const float* W2  = (const float*)d_in[3];
    const float* b2  = (const float*)d_in[4];
    const float* pw1 = (const float*)d_in[5];
    const float* pb1 = (const float*)d_in[6];
    const float* pw2 = (const float*)d_in[7];
    const float* pb2 = (const float*)d_in[8];
    const float* cw1 = (const float*)d_in[9];
    const float* cb1 = (const float*)d_in[10];
    const float* cw2 = (const float*)d_in[11];
    const float* cb2 = (const float*)d_in[12];
    const int* labels = (const int*)d_in[13];

    char* ws = (char*)d_ws;
    const size_t W_SZ   = (size_t)FDIM * FDIM * sizeof(_Float16);   // 1,179,648
    const size_t ACT_SZ = (size_t)NBATCH * NNODE * FDIM * sizeof(_Float16); // 132,120,576

    int*      kc   = (int*)(ws + 0);
    _Float16* W1t  = (_Float16*)(ws + (1 << 14));
    _Float16* W2t  = (_Float16*)(ws + (1 << 14) + W_SZ);
    _Float16* CW1t = (_Float16*)(ws + (1 << 14) + 2 * W_SZ);
    _Float16* PWt  = (_Float16*)(ws + (1 << 14) + 3 * W_SZ);
    _Float16* bufA = (_Float16*)(ws + (4ull << 20));               // 132 MB
    _Float16* bufB = (_Float16*)(ws + (4ull << 20) + ACT_SZ);      // 132 MB
    _Float16* gh   = (_Float16*)((char*)bufB + (32ull << 20));     // after Zh (22 MB)
    _Float16* Hh   = (_Float16*)((char*)bufB + (48ull << 20));

    const int M = NBATCH * NNODE;          // 86016 = 672*128
    const size_t total4 = (size_t)M * FDIM / 4;

    kcnt_kernel<<<(NBATCH + 255) / 256, 256, 0, stream>>>(labels, kc);
    tr_cvt<<<(FDIM * FDIM + 255) / 256, 256, 0, stream>>>(W1,  W1t,  FDIM, FDIM);
    tr_cvt<<<(FDIM * FDIM + 255) / 256, 256, 0, stream>>>(W2,  W2t,  FDIM, FDIM);
    tr_cvt<<<(FDIM * FDIM + 255) / 256, 256, 0, stream>>>(cw1, CW1t, FDIM, FDIM);
    build_pw<<<(128 * FDIM + 255) / 256, 256, 0, stream>>>(pw1, PWt);
    cvt_f32_h16<<<(unsigned)((total4 + 255) / 256), 256, 0, stream>>>(pooled, bufA, total4);

    dim3 gBig(M / 128, FDIM / 128);        // (672, 6)
    // layer 1
    wmma_gemm<false><<<gBig, 256, 0, stream>>>(bufA, W1t, bufB, nullptr, M, FDIM, FDIM);
    gcn_agg<<<NBATCH, 256, 0, stream>>>(bufB, b1, kc, bufA);
    // layer 2 + normalize
    wmma_gemm<false><<<gBig, 256, 0, stream>>>(bufA, W2t, bufB, nullptr, M, FDIM, FDIM);
    gcn_agg_norm<<<NBATCH, 256, 0, stream>>>(bufB, b2, kc, bufA);
    // pair projection: Z = X2 @ PW  (N=128)
    wmma_gemm<false><<<dim3(M / 128, 1), 256, 0, stream>>>(bufA, PWt, bufB, nullptr, M, 128, FDIM);
    // attention -> g (f16)
    attn_kernel<<<NBATCH, 256, 0, stream>>>(bufB, bufA, labels, pb1, pw2, pb2, gh);
    // classifier hidden: H = elu(g @ cw1 + cb1)
    wmma_gemm<true><<<dim3(NBATCH / 128, FDIM / 128), 256, 0, stream>>>(gh, CW1t, Hh, cb1, NBATCH, FDIM, FDIM);
    // head 768 -> 3
    final_kernel<<<NBATCH, 256, 0, stream>>>(Hh, cw2, cb2, (float*)d_out);
}
